// KernelSharingConv_34823594836064
// MI455X (gfx1250) — matrix operand order repaired and checked by reading, not compiled, not run on hardware
//
#include <hip/hip_runtime.h>
#include <math.h>

// ---------------------------------------------------------------------------
// CDNA5 (gfx1250) implicit-GEMM dilated conv + BN + exact GELU
//   5 branches (dilation 1,2,4,8,16), shared 3x3x64x64 kernel, NHWC.
//   Strategy: x -> bf16 once (fits in 192MB L2), weights prepacked into
//   WMMA B-fragment order, conv tile = 16 spatial x 64 Cout per wave32 using
//   v_wmma_f32_16x16x32_bf16 (K = Cin, contiguous in NHWC -> plain B128 loads).
// ---------------------------------------------------------------------------

typedef __attribute__((ext_vector_type(4)))  unsigned int v4u;
typedef __attribute__((ext_vector_type(2)))  unsigned int v2u;
typedef __attribute__((ext_vector_type(4)))  float       v4f;
typedef __attribute__((ext_vector_type(8)))  float       v8f;
typedef __attribute__((ext_vector_type(16))) __bf16      v16bf;

#define BN_EPS 1e-3f
#define X_ELEMS   33554432   // 8*256*256*64
#define OUT_PER_B 33554432   // per-branch output elements
#define WPACK_HALVES 36864   // 9 taps * 2 kchunks * 4 ntiles * 32 lanes * 16 halves

static __device__ __forceinline__ unsigned short f2bf(float f) {
  unsigned u = __float_as_uint(f);
  unsigned r = 0x7FFFu + ((u >> 16) & 1u);   // round-to-nearest-even
  return (unsigned short)((u + r) >> 16);
}

// -------- prepass 1: x (f32) -> bf16, 4 elements per thread ----------------
__global__ __launch_bounds__(256) void cvt_x_bf16(
    const float* __restrict__ x, unsigned short* __restrict__ xb, int n4) {
  int i = blockIdx.x * 256 + threadIdx.x;
  if (i >= n4) return;
  v4f f = ((const v4f*)x)[i];
  v2u p;
  p.x = (unsigned)f2bf(f.x) | ((unsigned)f2bf(f.y) << 16);
  p.y = (unsigned)f2bf(f.z) | ((unsigned)f2bf(f.w) << 16);
  ((v2u*)xb)[i] = p;
}

// -------- prepass 2: pack weights into per-lane-contiguous B fragments -----
// frag(t, kc, ct): 32 lanes x 16 halves. B layout (32x16 bf16):
//   lanes 0-15 : N=lane,    K = 0..7,16..23  (half h -> K = (h&7) + 16*(h>>3))
//   lanes 16-31: N=lane-16, K = 8..15,24..31 (half h -> K = 8 + (h&7) + 16*(h>>3))
__global__ __launch_bounds__(256) void pack_weights(
    const float* __restrict__ k, unsigned short* __restrict__ wp) {
  int idx = blockIdx.x * 256 + threadIdx.x;
  if (idx >= WPACK_HALVES) return;
  int h  = idx & 15;
  int l  = (idx >> 4) & 31;
  int ct = (idx >> 9) & 3;
  int kc = (idx >> 11) & 1;
  int t  = idx >> 12;                    // 0..8 = ky*3+kx
  int cin  = kc * 32 + ((l >> 4) << 3) + (h & 7) + ((h >> 3) << 4);
  int cout = (ct << 4) + (l & 15);
  wp[idx] = f2bf(k[(t * 64 + cin) * 64 + cout]);
}

// -------- main conv kernel: one wave32 -> 16(W) x 64(Cout) output tile -----
__global__ __launch_bounds__(256) void conv_bn_gelu_wmma(
    const unsigned short* __restrict__ xbf,
    const unsigned short* __restrict__ wp,
    const float* __restrict__ gamma, const float* __restrict__ beta,
    const float* __restrict__ mean,  const float* __restrict__ var,
    float* __restrict__ out) {
  const int lane = threadIdx.x & 31;
  const int wave = threadIdx.x >> 5;
  int wid = blockIdx.x * 8 + wave;          // 163840 tiles total
  const int xt = wid & 15;  wid >>= 4;      // 16 col tiles
  const int yy = wid & 255; wid >>= 8;      // row
  const int bb = wid & 7;   wid >>= 3;      // batch
  const int dd = wid;                       // branch 0..4
  const int dil = 1 << dd;                  // dilations 1,2,4,8,16
  const int x0  = xt << 4;

  // A fragment addressing (16x32 bf16): lanes 0-15 K in {0..7,16..23},
  // lanes 16-31 K in {8..15,24..31}; M = lane&15.
  const int mm = lane & 15;
  const int kb = (lane >> 4) << 3;          // 0 or 8

  const v4u* __restrict__ x4 = (const v4u*)xbf;
  const v4u* __restrict__ w4 = (const v4u*)wp;

  const v8f vzero8 = {0.f,0.f,0.f,0.f,0.f,0.f,0.f,0.f};
  const v4u vzero4 = {0u,0u,0u,0u};
  v8f acc[4];
#pragma unroll
  for (int i = 0; i < 4; ++i) acc[i] = vzero8;

#pragma unroll
  for (int t = 0; t < 9; ++t) {
    const int dy = t / 3 - 1, dx = t % 3 - 1;
    const int row = yy + dy * dil;
    const int col = x0 + mm + dx * dil;
    const bool valid = ((unsigned)row < 256u) & ((unsigned)col < 256u);
    const int pix = ((bb << 8) + row) * 256 + col;   // unused when !valid
#pragma unroll
    for (int kc = 0; kc < 2; ++kc) {
      union { v4u u[2]; v16bf v; } a;
      const int hoff = pix * 64 + kc * 32 + kb;      // in halves, 16B aligned
      if (valid) {
        a.u[0] = x4[(hoff >> 3)];        // K = kb .. kb+7
        a.u[1] = x4[(hoff >> 3) + 2];    // K = kb+16 .. kb+23
      } else {
        a.u[0] = vzero4;
        a.u[1] = vzero4;
      }
#pragma unroll
      for (int ct = 0; ct < 4; ++ct) {
        union { v4u u[2]; v16bf v; } b;
        const int fb = (((t * 2 + kc) * 4 + ct) * 32 + lane) * 2;  // v4u units
        b.u[0] = w4[fb];
        b.u[1] = w4[fb + 1];
        acc[ct] = __builtin_amdgcn_wmma_f32_16x16x32_bf16(
            false, a.v, false, b.v, (short)0, acc[ct], false, false);
      }
    }
  }

  // C/D layout: VGPR r holds M=r (lanes 0-15) / M=r+8 (lanes 16-31), N=lane&15
  const int mbase = (lane >> 4) << 3;
  const int nn    = lane & 15;
  const size_t outB = (size_t)dd * OUT_PER_B +
                      (size_t)(((bb << 8) + yy) * 256) * 64u;
#pragma unroll
  for (int ct = 0; ct < 4; ++ct) {
    const int c = (ct << 4) + nn;
    const int pc = dd * 64 + c;
    const float sc = gamma[pc] * rsqrtf(var[pc] + BN_EPS);
    const float mu = mean[pc];
    const float bt = beta[pc];
#pragma unroll
    for (int r = 0; r < 8; ++r) {
      const int m = r + mbase;
      float v = (acc[ct][r] - mu) * sc + bt;
      v = 0.5f * v * (1.0f + erff(v * 0.70710678118654752f));  // exact GELU
      out[outB + (size_t)(x0 + m) * 64u + c] = v;
    }
  }
}

// ---------------------------------------------------------------------------
extern "C" void kernel_launch(void* const* d_in, const int* in_sizes, int n_in,
                              void* d_out, int out_size, void* d_ws, size_t ws_size,
                              hipStream_t stream) {
  const float* x     = (const float*)d_in[0];
  const float* ker   = (const float*)d_in[1];
  const float* gamma = (const float*)d_in[2];
  const float* beta  = (const float*)d_in[3];
  const float* mmean = (const float*)d_in[4];
  const float* mvar  = (const float*)d_in[5];
  float* out = (float*)d_out;

  unsigned short* xb = (unsigned short*)d_ws;                       // 64 MiB bf16 x
  unsigned short* wp = (unsigned short*)((char*)d_ws + (size_t)X_ELEMS * 2); // 72 KiB

  // 1) x -> bf16 (lands in L2; re-read 45x by the conv at L2 bandwidth)
  cvt_x_bf16<<<(X_ELEMS / 4 + 255) / 256, 256, 0, stream>>>(x, xb, X_ELEMS / 4);
  // 2) weights -> WMMA B-fragment layout
  pack_weights<<<(WPACK_HALVES + 255) / 256, 256, 0, stream>>>(ker, wp);
  // 3) conv+BN+GELU: 5*8*256*16 = 163840 wave-tiles, 8 waves/block
  conv_bn_gelu_wmma<<<163840 / 8, 256, 0, stream>>>(
      xb, wp, gamma, beta, mmean, mvar, out);
}
